// MMUpdateMultimodal_17506286698525
// MI455X (gfx1250) — matrix operand order repaired and checked by reading, not compiled, not run on hardware
//
#include <hip/hip_runtime.h>
#include <hip/hip_bf16.h>

typedef __bf16 bf16;
typedef __attribute__((ext_vector_type(16))) __bf16 v16bf;
typedef __attribute__((ext_vector_type(8)))  __bf16 v8bf;
typedef __attribute__((ext_vector_type(4)))  __bf16 v4bf;
typedef __attribute__((ext_vector_type(8)))  float  v8f;
typedef __attribute__((ext_vector_type(4)))  unsigned int u32x4;
typedef __attribute__((ext_vector_type(8)))  int i32x8;
typedef __attribute__((ext_vector_type(4)))  int i32x4;

#define D_DIM   512
#define N_Q     784
#define M_MB    65536
#define QT      16                 // queries per workgroup tile (784 = 49*16)
#define MROWS   128                // memory rows per inner iteration (8 waves * 16)
#define CHUNK   1024               // memory rows per workgroup
#define NITER   (CHUNK / MROWS)    // 8
#define LDB     (D_DIM + 8)        // padded LDS row (bf16 elems) = TDM pad 16B/row
#define BTILE_B (MROWS * LDB * 2)  // 133,120 B per B buffer
#define A_B     (QT * LDB * 2)     // 16,640 B
#define NCHUNK  (M_MB / CHUNK)     // 64
#define CPQ     (NCHUNK * 4)       // candidates per query = 256
#define FLT_BIG 3.4e38f

// ---------------- sorted top-4 insert (ascending, d[0] smallest) --------------
__device__ __forceinline__ void ins4(float (&d)[4], int (&i)[4], float v, int id) {
  if (v < d[3]) {
    d[3] = v; i[3] = id;
    if (d[3] < d[2]) { float t=d[3]; d[3]=d[2]; d[2]=t; int u=i[3]; i[3]=i[2]; i[2]=u; }
    if (d[2] < d[1]) { float t=d[2]; d[2]=d[1]; d[1]=t; int u=i[2]; i[2]=i[1]; i[1]=u; }
    if (d[1] < d[0]) { float t=d[1]; d[1]=d[0]; d[0]=t; int u=i[1]; i[1]=i[0]; i[0]=u; }
  }
}

// ---- TDM: load one 128x512 bf16 tile (contiguous rows) into LDS with a 16B
//      pad after every 1KB row (rows land at 1040B stride = LDB elements).
//      D# per CDNA5 ISA (2D tensor, groups 2/3 zero).
__device__ __forceinline__ void tdm_load_tile(const bf16* gsrc, unsigned lds_off) {
  unsigned long long ga = (unsigned long long)(const void*)gsrc;
  u32x4 g0;
  g0[0] = 1u;                                        // count=1, user mode, no gather
  g0[1] = lds_off;                                   // lds_addr (bytes)
  g0[2] = (unsigned)(ga & 0xffffffffull);            // global_addr[31:0]
  g0[3] = (unsigned)((ga >> 32) & 0x1ffffffull)      // global_addr[56:32]
        | (2u << 30);                                // type = 2 ("image")
  i32x8 g1;
  g1[0] = (1 << 16)                                  // data_size = 1 -> 2 bytes
        | (1 << 20)                                  // pad_enable
        | (7 << 22)                                  // pad_interval = 256 DWORDs (1KB)
        | (3 << 25);                                 // pad_amount   = 4 DWORDs (16B)
  g1[1] = (D_DIM & 0xffff) << 16;                    // tensor_dim0[15:0] = 512
  g1[2] = (MROWS & 0xffff) << 16;                    // tensor_dim0[31:16]=0; tensor_dim1[15:0]=128
  g1[3] = (D_DIM & 0xffff) << 16;                    // tensor_dim1[31:16]=0; tile_dim0 = 512
  g1[4] = MROWS;                                     // tile_dim1 = 128; tile_dim2 = 0
  g1[5] = D_DIM;                                     // tensor_dim0_stride[31:0] = 512 elems
  g1[6] = 0;                                         // stride0 hi / dim1_stride lo
  g1[7] = 0;
  i32x4 z4 = {0, 0, 0, 0};                           // groups 2/3 unused (2D)
  i32x8 z8 = {0, 0, 0, 0, 0, 0, 0, 0};
  __builtin_amdgcn_tensor_load_to_lds(g0, g1, z4, z4, z8, 0);
}

// ---------------- kernel A: fused norms + fp32->bf16 bank conversion ----------
__global__ __launch_bounds__(256) void prep_kernel(const float* __restrict__ mb,
                                                   bf16* __restrict__ mbb,
                                                   float* __restrict__ m2) {
  int w = threadIdx.x >> 5, lane = threadIdx.x & 31;
  int row = blockIdx.x * 8 + w;                       // 8192 blocks * 8 waves
  const float4* r4 = (const float4*)(mb + (size_t)row * D_DIM);
  bf16* orow = mbb + (size_t)row * D_DIM;
  float s = 0.f;
  #pragma unroll
  for (int j = 0; j < 4; ++j) {
    float4 v = r4[lane + 32 * j];
    s += v.x*v.x + v.y*v.y + v.z*v.z + v.w*v.w;
    v4bf o = { (bf16)v.x, (bf16)v.y, (bf16)v.z, (bf16)v.w };
    *(v4bf*)(orow + 4 * (lane + 32 * j)) = o;
  }
  #pragma unroll
  for (int o = 16; o > 0; o >>= 1) s += __shfl_xor(s, o, 32);
  if (lane == 0) m2[row] = s;
}

// ---------------- kernel B: TDM double-buffered bf16 WMMA GEMM + fused top-4 --
__global__ __launch_bounds__(256) void gemm_topk_kernel(
    const float* __restrict__ q, const bf16* __restrict__ mbb,
    const float* __restrict__ m2, float* __restrict__ cand_d,
    int* __restrict__ cand_i) {
  extern __shared__ char smem[];
  bf16* As  = (bf16*)smem;                            // 16,640 B
  bf16* Bs0 = (bf16*)(smem + A_B);                    // 133,120 B
  bf16* Bs1 = (bf16*)(smem + A_B + BTILE_B);          // 133,120 B  (total 282,880)
  // staging overlays Bs0 after all compute is done:
  float* sd  = (float*)(smem + A_B);                  // 8w*16q*16l*4 = 32 KB
  int*   si  = (int*)  (smem + A_B + 32768);          // 32 KB
  float* sd2 = (float*)(smem + A_B + 65536);          // 16q*8w*4 = 2 KB
  int*   si2 = (int*)  (smem + A_B + 65536 + 2048);

  const int t    = threadIdx.x;
  const int w    = t >> 5, lane = t & 31;
  const int kh   = lane >> 4;         // K-half selector (ISA A/B lane layout)
  const int l16  = lane & 15;
  const int qt   = blockIdx.x;        // 0..48 query tile
  const int ck   = blockIdx.y;        // 0..63 memory chunk
  const int m0   = ck * CHUNK;

  const unsigned bs_off[2] = {
    (unsigned)(unsigned long long)(void*)Bs0,
    (unsigned)(unsigned long long)(void*)Bs1 };

  // A tile: 16x512 fp32 -> bf16 LDS (2048 float4, 8 per thread)
  const float4* q4 = (const float4*)(q + (size_t)qt * QT * D_DIM);
  #pragma unroll
  for (int j = 0; j < 8; ++j) {
    int f = t + 256 * j;
    float4 v = q4[f];
    int row = f >> 7, col = (f & 127) << 2;
    v4bf o = { (bf16)v.x, (bf16)v.y, (bf16)v.z, (bf16)v.w };
    *(v4bf*)(As + row * LDB + col) = o;
  }

  float bd[8][4]; int bi[8][4];       // per-lane sorted top-4 per D-matrix slot
  #pragma unroll
  for (int r = 0; r < 8; ++r)
    #pragma unroll
    for (int k = 0; k < 4; ++k) { bd[r][k] = FLT_BIG; bi[r][k] = 0; }

  // preload first B tile via TDM (wave 0 issues; EXEC ignored by tensor ops)
  if (t == 0) tdm_load_tile(mbb + (size_t)m0 * D_DIM, bs_off[0]);

  for (int it = 0; it < NITER; ++it) {
    const int p = it & 1;
    if (t == 0) __builtin_amdgcn_s_wait_tensorcnt(0);   // tile 'it' landed
    __syncthreads();        // data visible to all waves; all done with buf[1-p]
    if (t == 0 && it + 1 < NITER)                       // DMA next tile while computing
      tdm_load_tile(mbb + (size_t)(m0 + (it + 1) * MROWS) * D_DIM, bs_off[1 - p]);

    const bf16* Bs  = p ? Bs1 : Bs0;
    const int  mcol = m0 + it * MROWS + w * 16 + l16;
    const float m2v = m2[mcol];

    v8f acc = {0.f, 0.f, 0.f, 0.f, 0.f, 0.f, 0.f, 0.f};
    const bf16* arow = As + l16 * LDB;                  // query row = lane%16
    const bf16* brow = Bs + (w * 16 + l16) * LDB;       // B column  = mem row
    #pragma unroll
    for (int kk = 0; kk < D_DIM; kk += 32) {
      // A 16x32 bf16 layout: lanes 0-15 hold K 0-7 & 16-23; lanes 16-31 hold 8-15 & 24-31
      v8bf alo = *(const v8bf*)(arow + kk + kh * 8);
      v8bf ahi = *(const v8bf*)(arow + kk + kh * 8 + 16);
      v16bf a = __builtin_shufflevector(alo, ahi, 0,1,2,3,4,5,6,7,8,9,10,11,12,13,14,15);
      // B 32x16 bf16 layout: lanes 0-15 hold K 0-7 & 16-23; lanes 16-31 hold 8-15 & 24-31
      v8bf blo = *(const v8bf*)(brow + kk + kh * 8);
      v8bf bhi = *(const v8bf*)(brow + kk + kh * 8 + 16);
      v16bf b = __builtin_shufflevector(blo, bhi, 0,1,2,3,4,5,6,7,8,9,10,11,12,13,14,15);
      acc = __builtin_amdgcn_wmma_f32_16x16x32_bf16(false, a, false, b,
                                                    (short)0, acc, false, false);
    }
    // d2 ranking value: m2 - 2*dot (q2 constant per query row -> dropped)
    #pragma unroll
    for (int r = 0; r < 8; ++r) {
      float v = m2v - 2.0f * acc[r];
      ins4(bd[r], bi[r], v, mcol);
    }
  }

  __syncthreads();                       // all compute done: overlay staging on Bs0
  #pragma unroll
  for (int r = 0; r < 8; ++r) {
    int qrow = r + 8 * kh;               // D layout: slot r, lane half -> query row
    int base = w * 1024 + qrow * 64 + l16 * 4;
    #pragma unroll
    for (int k = 0; k < 4; ++k) { sd[base + k] = bd[r][k]; si[base + k] = bi[r][k]; }
  }
  __syncthreads();
  if (lane < 16) {                       // wave-local merge: 64 -> 4 per (wave, query)
    float fd[4] = {FLT_BIG, FLT_BIG, FLT_BIG, FLT_BIG};
    int   fi[4] = {0, 0, 0, 0};
    int base = w * 1024 + lane * 64;
    for (int s = 0; s < 64; ++s) ins4(fd, fi, sd[base + s], si[base + s]);
    int b2 = (lane * 8 + w) * 4;
    #pragma unroll
    for (int k = 0; k < 4; ++k) { sd2[b2 + k] = fd[k]; si2[b2 + k] = fi[k]; }
  }
  __syncthreads();
  if (w == 0 && lane < 16) {             // cross-wave merge: 32 -> 4 per query
    float fd[4] = {FLT_BIG, FLT_BIG, FLT_BIG, FLT_BIG};
    int   fi[4] = {0, 0, 0, 0};
    int base = lane * 32;
    for (int s = 0; s < 32; ++s) ins4(fd, fi, sd2[base + s], si2[base + s]);
    int n = qt * QT + lane;
    size_t cb = ((size_t)n * NCHUNK + ck) * 4;
    #pragma unroll
    for (int k = 0; k < 4; ++k) { cand_d[cb + k] = fd[k]; cand_i[cb + k] = fi[k]; }
  }
}

// ---------------- kernel C: merge 256 candidates/query -> final top-4 ---------
__global__ __launch_bounds__(256) void merge_topk_kernel(
    const float* __restrict__ cand_d, const int* __restrict__ cand_i,
    int* __restrict__ topk) {
  __shared__ float ldd[8 * 32 * 4];
  __shared__ int   ldi[8 * 32 * 4];
  int w = threadIdx.x >> 5, lane = threadIdx.x & 31;
  int n = blockIdx.x * 8 + w;                          // 98 * 8 = 784
  float d[4] = {FLT_BIG, FLT_BIG, FLT_BIG, FLT_BIG};
  int   id[4] = {0, 0, 0, 0};
  size_t base = (size_t)n * CPQ;
  #pragma unroll
  for (int j = 0; j < CPQ / 32; ++j) {
    int s = lane + 32 * j;
    ins4(d, id, cand_d[base + s], cand_i[base + s]);
  }
  int lb = (w * 32 + lane) * 4;
  #pragma unroll
  for (int k = 0; k < 4; ++k) { ldd[lb + k] = d[k]; ldi[lb + k] = id[k]; }
  __syncthreads();
  if (lane == 0) {
    float fd[4] = {FLT_BIG, FLT_BIG, FLT_BIG, FLT_BIG};
    int   fi[4] = {0, 0, 0, 0};
    int wb = w * 128;
    for (int s = 0; s < 128; ++s) ins4(fd, fi, ldd[wb + s], ldi[wb + s]);
    #pragma unroll
    for (int k = 0; k < 4; ++k) topk[n * 4 + k] = fi[k];
  }
}

// ---------------- kernel D: gather neighbors, mean, build combined -----------
__global__ __launch_bounds__(128) void gather_kernel(
    const float* __restrict__ q, const float* __restrict__ mb,
    const int* __restrict__ topk, float* __restrict__ comb) {
  int n = blockIdx.x;
  const float* r0 = mb + (size_t)topk[n * 4 + 0] * D_DIM;
  const float* r1 = mb + (size_t)topk[n * 4 + 1] * D_DIM;
  const float* r2 = mb + (size_t)topk[n * 4 + 2] * D_DIM;
  const float* r3 = mb + (size_t)topk[n * 4 + 3] * D_DIM;
  for (int c = threadIdx.x; c < 2 * D_DIM; c += 128) {
    float v;
    if (c < D_DIM) v = q[(size_t)n * D_DIM + c];
    else { int d = c - D_DIM; v = 0.25f * (r0[d] + r1[d] + r2[d] + r3[d]); }
    comb[(size_t)n * (2 * D_DIM) + c] = v;
  }
}

// ---------------- kernel E: 3x3 conv (1024 -> 1 channels) + bias -------------
__global__ __launch_bounds__(256) void conv_kernel(
    const float* __restrict__ comb, const float* __restrict__ wgt,
    const float* __restrict__ bias, float* __restrict__ out) {
  int w = threadIdx.x >> 5, lane = threadIdx.x & 31;
  int p = blockIdx.x * 8 + w;                          // 98 * 8 = 784 pixels
  int y = p / 28, x = p % 28;
  float s = 0.f;
  #pragma unroll
  for (int ky = 0; ky < 3; ++ky) {
    int py = y + ky - 1;
    if (py < 0 || py >= 28) continue;
    #pragma unroll
    for (int kx = 0; kx < 3; ++kx) {
      int px = x + kx - 1;
      if (px < 0 || px >= 28) continue;
      const float* crow = comb + (size_t)(py * 28 + px) * (2 * D_DIM);
      for (int cc = 0; cc < 32; ++cc) {
        int c = lane + 32 * cc;
        s += wgt[(c * 3 + ky) * 3 + kx] * crow[c];     // OIHW: [0][c][ky][kx]
      }
    }
  }
  #pragma unroll
  for (int o = 16; o > 0; o >>= 1) s += __shfl_xor(s, o, 32);
  if (lane == 0) out[p] = s + bias[0];
}

extern "C" void kernel_launch(void* const* d_in, const int* in_sizes, int n_in,
                              void* d_out, int out_size, void* d_ws, size_t ws_size,
                              hipStream_t stream) {
  const float* q    = (const float*)d_in[0];   // [784, 512]
  const float* mb   = (const float*)d_in[1];   // [65536, 512]
  const float* wgt  = (const float*)d_in[2];   // [1, 1024, 3, 3]
  const float* bias = (const float*)d_in[3];   // [1]
  float* out = (float*)d_out;                  // [1,1,28,28] = 784

  char* ws = (char*)d_ws;
  bf16*  mbb    = (bf16*)ws;   ws += (size_t)M_MB * D_DIM * 2;         // 64 MB bf16 bank
  float* m2     = (float*)ws;  ws += (size_t)M_MB * 4;                 // 256 KB
  float* cand_d = (float*)ws;  ws += (size_t)N_Q * CPQ * 4;            // 784 KB
  int*   cand_i = (int*)ws;    ws += (size_t)N_Q * CPQ * 4;            // 784 KB
  int*   topk   = (int*)ws;    ws += 12800;                            // 784*4 ints, padded
  float* comb   = (float*)ws;                                          // 784*1024 f32

  const int smemB = A_B + 2 * BTILE_B;                                 // 282,880 B
  (void)hipFuncSetAttribute(reinterpret_cast<const void*>(gemm_topk_kernel),
                            hipFuncAttributeMaxDynamicSharedMemorySize, smemB);

  prep_kernel<<<M_MB / 8, 256, 0, stream>>>(mb, mbb, m2);
  gemm_topk_kernel<<<dim3(N_Q / QT, NCHUNK), 256, smemB, stream>>>(q, mbb, m2,
                                                                   cand_d, cand_i);
  merge_topk_kernel<<<N_Q / 8, 256, 0, stream>>>(cand_d, cand_i, topk);
  gather_kernel<<<N_Q, 128, 0, stream>>>(q, mb, topk, comb);
  conv_kernel<<<N_Q / 8, 256, 0, stream>>>(comb, wgt, bias, out);
}
